// Block_24292335026759
// MI455X (gfx1250) — compile-verified, matrix operand order checked by reading, not run on hardware
//
#include <hip/hip_runtime.h>
#include <hip/hip_bf16.h>
#include <math.h>

// ---------------------------------------------------------------------------
// MI455X (gfx1250, wave32) fused cross-attention block.
// - all GEMM math on v_wmma_f32_16x16x32_f16
// - A tiles staged to LDS with global_load_async_to_lds_b128 (+ s_wait_asynccnt)
// - weights pre-packed to f16 [N][K] so every fragment load is 128-bit
// ---------------------------------------------------------------------------

typedef __attribute__((ext_vector_type(16))) _Float16 v16h;
typedef __attribute__((ext_vector_type(8)))  _Float16 v8h;
typedef __attribute__((ext_vector_type(8)))  float    v8f;

#define DIMC 512
#define HIDC 128
#define MLPC 1024
#define SEQC 2048
#define NBATCH 8

__device__ __forceinline__ v8f wmma_f16(v16h a, v16h b, v8f c) {
  return __builtin_amdgcn_wmma_f32_16x16x32_f16(
      false, a, false, b, (short)0, c, false, false);
}

// f16 fragment (16x32 A or 32x16 B) from a row pointer already offset by kk:
// the ISA 7.12.2 16-bit layout is two contiguous 8-element runs per lane,
// at +8*half and +16+8*half  ->  two 16-byte loads.
__device__ __forceinline__ v16h frag_h(const _Float16* __restrict__ row, int half) {
  const v8h lo = *(const v8h*)(row + 8 * half);
  const v8h hi = *(const v8h*)(row + 16 + 8 * half);
  v16h r;
#pragma unroll
  for (int i = 0; i < 8; ++i) { r[i] = lo[i]; r[8 + i] = hi[i]; }
  return r;
}

// Same, but source data is f32 (converted to f16 in registers).
__device__ __forceinline__ v16h frag_f(const float* __restrict__ row, int half) {
  v16h r;
#pragma unroll
  for (int i = 0; i < 8; ++i) {
    r[i]     = (_Float16)row[8 * half + i];
    r[8 + i] = (_Float16)row[16 + 8 * half + i];
  }
  return r;
}

__device__ __forceinline__ float redmax16(float v) {
#pragma unroll
  for (int m = 1; m < 16; m <<= 1) v = fmaxf(v, __shfl_xor(v, m, 32));
  return v;
}
__device__ __forceinline__ float redsum16(float v) {
#pragma unroll
  for (int m = 1; m < 16; m <<= 1) v += __shfl_xor(v, m, 32);
  return v;
}

__device__ __forceinline__ float gelu_exact(float x) {
  return 0.5f * x * (1.0f + erff(x * 0.7071067811865476f));
}

// Async-copy a contiguous global block into LDS (16B per lane per issue),
// then drain ASYNCcnt.  CDNA5 GLOBAL_LOAD_ASYNC_TO_LDS_B128 path.
__device__ __forceinline__ void async_stage(const char* __restrict__ g,
                                            char* lds, int bytes, int tid) {
  const unsigned ldsBase = (unsigned)(size_t)(void*)lds;  // low 32b = LDS addr
  for (int off = tid * 16; off < bytes; off += 256 * 16) {
    const unsigned l = ldsBase + (unsigned)off;
    const char* p = g + off;
    asm volatile("global_load_async_to_lds_b128 %0, %1, off"
                 :: "v"(l), "v"(p) : "memory");
  }
  asm volatile("s_wait_asynccnt 0x0" ::: "memory");
}

// ---------------------------------------------------------------------------
// Pack f32 weights W[K][N] -> f16 Wt[N][K] (transposed) so B fragments are
// contiguous 128-bit loads.
// ---------------------------------------------------------------------------
__global__ __launch_bounds__(256) void pack_wt_kernel(
    const float* __restrict__ W, _Float16* __restrict__ Wt, int K, int N) {
  const int idx = blockIdx.x * 256 + threadIdx.x;
  if (idx < K * N) {
    const int k = idx / N, n = idx % N;
    Wt[(size_t)n * K + k] = (_Float16)W[idx];
  }
}

// ---------------------------------------------------------------------------
// Generic 16-row-tile WMMA GEMM:  out[M,N] = epilogue(A[M,K] @ W[K,N])
//   A staged to LDS via async-to-LDS (f32 raw if !aHalf, f16 if aHalf)
//   Wt: f16 [N][K] packed weights
//   mode 0: store f16 row-major
//   mode 1: store f16 transposed per batch  ([b][N][rowsPerBatch])
//   mode 2: +bias, exact GELU, store f16
//   mode 3: +bias, +resid (f32), store f32 (final output)
// ---------------------------------------------------------------------------
__global__ __launch_bounds__(256) void gemm_wmma_kernel(
    const void* __restrict__ Ap, int aHalf,
    const _Float16* __restrict__ Wt,
    const float* __restrict__ bias,
    const float* __restrict__ resid,
    void* __restrict__ outp,
    int K, int N, int mode, int rowsPerBatch) {
  extern __shared__ char AsB[];  // 16 x K  (f32 or f16)
  const int tid  = threadIdx.x;
  const int wave = tid >> 5, lane = tid & 31;
  const int half = lane >> 4, col = lane & 15, mrow = lane & 15;
  const size_t rowBase = (size_t)blockIdx.x * 16;

  // Stage the contiguous 16-row A tile into LDS asynchronously.
  const int esz    = aHalf ? 2 : 4;
  const int abytes = 16 * K * esz;
  async_stage((const char*)Ap + rowBase * K * esz, AsB, abytes, tid);
  __syncthreads();

  const _Float16* AsH = (const _Float16*)AsB;
  const float*    AsF = (const float*)AsB;

  for (int n0 = wave * 16; n0 < N; n0 += 8 * 16) {
    const _Float16* wRow = Wt + (size_t)(n0 + col) * K;
    v8f acc = {};
    if (aHalf) {
      for (int kk = 0; kk < K; kk += 32) {
        if (kk + 64 < K)  // gfx1250 global_prefetch_b8
          __builtin_prefetch(wRow + kk + 64, 0, 1);
        acc = wmma_f16(frag_h(AsH + mrow * K + kk, half),
                       frag_h(wRow + kk, half), acc);
      }
    } else {
      for (int kk = 0; kk < K; kk += 32) {
        if (kk + 64 < K)
          __builtin_prefetch(wRow + kk + 64, 0, 1);
        acc = wmma_f16(frag_f(AsF + mrow * K + kk, half),
                       frag_h(wRow + kk, half), acc);
      }
    }
#pragma unroll
    for (int r = 0; r < 8; ++r) {
      const int row = r + 8 * half;
      const int gc  = n0 + col;
      const size_t gr = rowBase + row;
      float v = acc[r];
      if (bias) v += bias[gc];
      if (mode == 2) v = gelu_exact(v);
      if (mode == 3) {
        float* oF = (float*)outp;
        oF[gr * N + gc] = v + resid[gr * N + gc];
      } else if (mode == 1) {
        _Float16* oH = (_Float16*)outp;
        const size_t bidx = gr / rowsPerBatch;
        const size_t rin  = gr % rowsPerBatch;
        oH[(bidx * N + gc) * rowsPerBatch + rin] = (_Float16)v;
      } else {
        _Float16* oH = (_Float16*)outp;
        oH[gr * N + gc] = (_Float16)v;
      }
    }
  }
}

// ---------------------------------------------------------------------------
// Flash attention (online softmax, 32 keys/step) + residual + LayerNorm.
// One block = 16 queries of one batch; 8 waves x 64 channels = DIM.
//   q:[b][S][HID] f16   k:[b][S][HID] f16 (row-major, indexed as B by key row)
//   vT:[b][DIM][S] f16  x:[b][S][DIM] f32
// Per step: 8 WMMA for S (2 key sub-tiles x 4 K-steps over HID) and
//           4 WMMA for P@V (K=32 keys, 4 channel tiles).
// ---------------------------------------------------------------------------
__global__ __launch_bounds__(256) void attn_ln_kernel(
    const _Float16* __restrict__ q,
    const _Float16* __restrict__ kmat,
    const _Float16* __restrict__ vT,
    const float* __restrict__ x,
    const float* __restrict__ gamma,
    const float* __restrict__ beta,
    float* __restrict__ xAttn,
    _Float16* __restrict__ hnorm) {
  __shared__ _Float16 smemP[8][16 * 32];   // per-wave 16x32 P tile
  __shared__ float    smemXO[16][DIMC];
  __shared__ float    sred1[16][16];
  __shared__ float    sred2[16][16];

  const int tid  = threadIdx.x;
  const int wave = tid >> 5, lane = tid & 31;
  const int half = lane >> 4, col = lane & 15, mrow = lane & 15;
  const int qTiles  = SEQC / 16;
  const int batch   = blockIdx.x / qTiles;
  const int qt      = blockIdx.x % qTiles;
  const int rowBase = qt * 16;

  const _Float16* qB  = q    + ((size_t)batch * SEQC + rowBase) * HIDC;
  const _Float16* kB  = kmat + (size_t)batch * SEQC * HIDC;
  const _Float16* vTB = vT   + (size_t)batch * DIMC * SEQC;

  v16h qf[4];
#pragma unroll
  for (int t = 0; t < 4; ++t) qf[t] = frag_h(qB + mrow * HIDC + t * 32, half);

  v8f accO[4];
  {
    v8f z = {};
#pragma unroll
    for (int t = 0; t < 4; ++t) accO[t] = z;
  }
  float rm[8], rl[8];
#pragma unroll
  for (int r = 0; r < 8; ++r) { rm[r] = -1e30f; rl[r] = 0.0f; }

  const float scale  = 0.08838834764831845f;  // HID^-0.5
  const int   chBase = wave * 64;

  for (int kt = 0; kt < SEQC / 32; ++kt) {
    const int keyBase = kt * 32;
    const _Float16* kRow0 = kB + (size_t)(keyBase + col) * HIDC;
    const _Float16* kRow1 = kB + (size_t)(keyBase + 16 + col) * HIDC;
    v8f s0 = {}, s1 = {};
#pragma unroll
    for (int t = 0; t < 4; ++t) {
      s0 = wmma_f16(qf[t], frag_h(kRow0 + t * 32, half), s0);
      s1 = wmma_f16(qf[t], frag_h(kRow1 + t * 32, half), s1);
    }
    // Online softmax over 32 keys (half-wave reductions match C row striping).
    float cf[8];
#pragma unroll
    for (int r = 0; r < 8; ++r) {
      const float sv0 = s0[r] * scale;
      const float sv1 = s1[r] * scale;
      const float tm  = redmax16(fmaxf(sv0, sv1));
      const float nm  = fmaxf(rm[r], tm);
      cf[r] = __expf(rm[r] - nm);
      rm[r] = nm;
      const float pv0 = __expf(sv0 - nm);
      const float pv1 = __expf(sv1 - nm);
      rl[r] = rl[r] * cf[r] + redsum16(pv0 + pv1);
      const int row = r + 8 * half;
      smemP[wave][row * 32 + col]      = (_Float16)pv0;
      smemP[wave][row * 32 + 16 + col] = (_Float16)pv1;
    }
#pragma unroll
    for (int t = 0; t < 4; ++t)
#pragma unroll
      for (int r = 0; r < 8; ++r) accO[t][r] *= cf[r];

    // P is exactly a 16x32 A fragment; V columns are contiguous in vT.
    v16h pf = frag_h(&smemP[wave][mrow * 32], half);
#pragma unroll
    for (int t = 0; t < 4; ++t) {
      const _Float16* vRow = vTB + (size_t)(chBase + t * 16 + col) * SEQC + keyBase;
      accO[t] = wmma_f16(pf, frag_h(vRow, half), accO[t]);
    }
  }

  // Normalize, add residual, stash tile for LayerNorm.
#pragma unroll
  for (int t = 0; t < 4; ++t) {
#pragma unroll
    for (int r = 0; r < 8; ++r) {
      const int row = r + 8 * half;
      const int ch  = chBase + t * 16 + col;
      const float o = accO[t][r] / rl[r];
      const size_t gidx = ((size_t)batch * SEQC + rowBase + row) * DIMC + ch;
      const float xo = x[gidx] + o;
      xAttn[gidx] = xo;
      smemXO[row][ch] = xo;
    }
  }
  __syncthreads();

  // LayerNorm over the 16x512 tile: 16 lanes per row.
  const int lrow = tid >> 4;
  const int sub  = tid & 15;
  float s1s = 0.0f, s2s = 0.0f;
  for (int c = sub; c < DIMC; c += 16) {
    const float vv = smemXO[lrow][c];
    s1s += vv; s2s += vv * vv;
  }
  sred1[lrow][sub] = s1s;
  sred2[lrow][sub] = s2s;
  __syncthreads();
  float mu = 0.0f, m2 = 0.0f;
#pragma unroll
  for (int i = 0; i < 16; ++i) { mu += sred1[lrow][i]; m2 += sred2[lrow][i]; }
  mu *= (1.0f / DIMC);
  const float var  = m2 * (1.0f / DIMC) - mu * mu;
  const float rstd = rsqrtf(var + 1e-5f);
  for (int c = sub; c < DIMC; c += 16) {
    const float hv = (smemXO[lrow][c] - mu) * rstd * gamma[c] + beta[c];
    hnorm[((size_t)batch * SEQC + rowBase + lrow) * DIMC + c] = (_Float16)hv;
  }
}

// ---------------------------------------------------------------------------
extern "C" void kernel_launch(void* const* d_in, const int* in_sizes, int n_in,
                              void* d_out, int out_size, void* d_ws, size_t ws_size,
                              hipStream_t stream) {
  const float* x     = (const float*)d_in[0];
  const float* y     = (const float*)d_in[1];
  const float* Wq    = (const float*)d_in[2];
  const float* Wk    = (const float*)d_in[3];
  const float* Wv    = (const float*)d_in[4];
  const float* gamma = (const float*)d_in[5];
  const float* beta  = (const float*)d_in[6];
  const float* W1    = (const float*)d_in[7];
  const float* b1    = (const float*)d_in[8];
  const float* W2    = (const float*)d_in[9];
  const float* b2    = (const float*)d_in[10];
  (void)in_sizes; (void)n_in; (void)out_size; (void)ws_size;

  const size_t M = (size_t)NBATCH * SEQC;  // 16384 rows

  // Workspace carve-up.
  char* ws = (char*)d_ws;
  _Float16* qh   = (_Float16*)ws; ws += M * HIDC * sizeof(_Float16);
  _Float16* kh   = (_Float16*)ws; ws += M * HIDC * sizeof(_Float16);
  _Float16* vTh  = (_Float16*)ws; ws += M * DIMC * sizeof(_Float16);
  float*    xAtt = (float*)ws;    ws += M * DIMC * sizeof(float);
  _Float16* hn   = (_Float16*)ws; ws += M * DIMC * sizeof(_Float16);
  _Float16* gmid = (_Float16*)ws; ws += M * MLPC * sizeof(_Float16);
  _Float16* WtQ  = (_Float16*)ws; ws += (size_t)DIMC * HIDC * sizeof(_Float16);
  _Float16* WtK  = (_Float16*)ws; ws += (size_t)DIMC * HIDC * sizeof(_Float16);
  _Float16* WtV  = (_Float16*)ws; ws += (size_t)DIMC * DIMC * sizeof(_Float16);
  _Float16* Wt1  = (_Float16*)ws; ws += (size_t)DIMC * MLPC * sizeof(_Float16);
  _Float16* Wt2  = (_Float16*)ws; ws += (size_t)MLPC * DIMC * sizeof(_Float16);

  // Pack weights -> f16 [N][K].
  pack_wt_kernel<<<(DIMC * HIDC + 255) / 256, 256, 0, stream>>>(Wq, WtQ, DIMC, HIDC);
  pack_wt_kernel<<<(DIMC * HIDC + 255) / 256, 256, 0, stream>>>(Wk, WtK, DIMC, HIDC);
  pack_wt_kernel<<<(DIMC * DIMC + 255) / 256, 256, 0, stream>>>(Wv, WtV, DIMC, DIMC);
  pack_wt_kernel<<<(DIMC * MLPC + 255) / 256, 256, 0, stream>>>(W1, Wt1, DIMC, MLPC);
  pack_wt_kernel<<<(MLPC * DIMC + 255) / 256, 256, 0, stream>>>(W2, Wt2, MLPC, DIMC);

  const int blocks = (int)(M / 16);  // 1024
  const size_t ldsF32K512  = (size_t)16 * DIMC * sizeof(float);      // 32 KB
  const size_t ldsF16K512  = (size_t)16 * DIMC * sizeof(_Float16);   // 16 KB
  const size_t ldsF16K1024 = (size_t)16 * MLPC * sizeof(_Float16);   // 32 KB

  // q = x@Wq (f16), k = y@Wk (f16 row-major), vT = (y@Wv)^T (f16)
  gemm_wmma_kernel<<<blocks, 256, ldsF32K512, stream>>>(
      x, 0, WtQ, nullptr, nullptr, qh, DIMC, HIDC, 0, SEQC);
  gemm_wmma_kernel<<<blocks, 256, ldsF32K512, stream>>>(
      y, 0, WtK, nullptr, nullptr, kh, DIMC, HIDC, 0, SEQC);
  gemm_wmma_kernel<<<blocks, 256, ldsF32K512, stream>>>(
      y, 0, WtV, nullptr, nullptr, vTh, DIMC, DIMC, 1, SEQC);

  // Flash attention + residual + LayerNorm (fused).
  attn_ln_kernel<<<blocks, 256, 0, stream>>>(
      qh, kh, vTh, x, gamma, beta, xAtt, hn);

  // MLP: gelu(hnorm @ W1 + b1) @ W2 + b2 + xAttn -> d_out (f32)
  gemm_wmma_kernel<<<blocks, 256, ldsF16K512, stream>>>(
      hn, 1, Wt1, b1, nullptr, gmid, DIMC, MLPC, 2, SEQC);
  gemm_wmma_kernel<<<blocks, 256, ldsF16K1024, stream>>>(
      gmid, 1, Wt2, b2, xAtt, d_out, MLPC, DIMC, 3, SEQC);
}
